// GNNModel_21131239096771
// MI455X (gfx1250) — matrix operand order repaired and checked by reading, not compiled, not run on hardware
//
#include <hip/hip_runtime.h>
#include <cstdint>
#include <cstddef>

#define Nn 100000
#define EH 64
#define RBF_LOW_F  0.005f
#define RBF_HIGH_F 0.2f

typedef _Float16 h16;
typedef __attribute__((ext_vector_type(16))) _Float16 v16h;
typedef __attribute__((ext_vector_type(8)))  float    v8f;
typedef __attribute__((ext_vector_type(4)))  float    v4f;

union UA { v16h v; h16 h[16]; v4f f4[2]; };
union U8 { h16 h[8]; v4f f4; };
union UC { v8f v; float f[8]; };

// fragment table offsets (units: fragments of 512 halves = 1 KB each)
#define FRAG_W0  0    // edge W0 64x64 : 4 coltiles x 2 ktiles
#define FRAG_W1  8
#define FRAG_W2  16   // edge W2 64x8  : 1 coltile (zero-padded) x 2 ktiles
#define FRAG_MP  18   // 4 layers x (2 coltiles x 8 ktiles)
#define FRAG_FC0 82   // 32x128 : 8 coltiles x 1 ktile
#define FRAG_FC1 90   // 128x128: 8 x 4
#define FRAG_FC2 122
#define FRAG_OUT 154  // 128x32 : 2 x 4
#define NFRAGS   162

__device__ __forceinline__ void wbar() { __builtin_amdgcn_wave_barrier(); }

__device__ __forceinline__ v8f wmma16(v16h a, v16h b, v8f c) {
  return __builtin_amdgcn_wmma_f32_16x16x32_f16(false, a, false, b, (short)0, c,
                                                false, false);
}

// fast transcendentals on the hardware v_exp/v_log path
__device__ __forceinline__ float fast_tanh(float x) {
  float xc = fminf(fmaxf(x, -15.f), 15.f);
  float e = __expf(2.f * xc);
  return (e - 1.f) / (e + 1.f);
}
__device__ __forceinline__ float fast_softplus(float x) {
  return (x > 20.f) ? x : __logf(1.f + __expf(x));
}

// B fragment: pre-swizzled in global ws; lane's 16 halves are contiguous (32B)
__device__ __forceinline__ v16h load_bfrag(const h16* fb, int f, int lane) {
  UA u;
  const h16* p = fb + (size_t)f * 512 + (size_t)lane * 16;
  u.f4[0] = *(const v4f*)p;
  u.f4[1] = *(const v4f*)(p + 8);
  return u.v;
}

// A fragment from an LDS row-major f16 tile (row = lane&15; interleaved K runs)
__device__ __forceinline__ v16h load_afrag(const h16* X, int stride, int lane, int kt) {
  int r = lane & 15, hh = lane >> 4;
  int k0 = kt * 32 + hh * 8;
  UA u;
  u.f4[0] = *(const v4f*)(X + r * stride + k0);
  u.f4[1] = *(const v4f*)(X + r * stride + k0 + 16);
  return u.v;
}

// ---------------- prep: weights -> f16 B fragments ----------------
__global__ void prep_frags(const float* w0, const float* w1, const float* w2,
                           const float* mp0, const float* mp1, const float* mp2,
                           const float* mp3, const float* fc0, const float* fc1,
                           const float* fc2, const float* ow, h16* fb) {
  int t = blockIdx.x * blockDim.x + threadIdx.x;
  if (t >= NFRAGS * 32) return;
  int f = t >> 5, lane = t & 31;
  const float* W; int ldN, nvalid, ct, kt;
  if (f < FRAG_W1)        { W = w0; ldN = 64;  nvalid = 64;  int g = f - FRAG_W0;  ct = g >> 1; kt = g & 1; }
  else if (f < FRAG_W2)   { W = w1; ldN = 64;  nvalid = 64;  int g = f - FRAG_W1;  ct = g >> 1; kt = g & 1; }
  else if (f < FRAG_MP)   { W = w2; ldN = 8;   nvalid = 8;   ct = 0; kt = f - FRAG_W2; }
  else if (f < FRAG_FC0)  { int g = f - FRAG_MP; int L = g >> 4; int h = g & 15;
                            ct = h >> 3; kt = h & 7; ldN = 32; nvalid = 32;
                            W = (L == 0) ? mp0 : (L == 1) ? mp1 : (L == 2) ? mp2 : mp3; }
  else if (f < FRAG_FC1)  { W = fc0; ldN = 128; nvalid = 128; ct = f - FRAG_FC0; kt = 0; }
  else if (f < FRAG_FC2)  { int g = f - FRAG_FC1; W = fc1; ldN = 128; nvalid = 128; ct = g >> 2; kt = g & 3; }
  else if (f < FRAG_OUT)  { int g = f - FRAG_FC2; W = fc2; ldN = 128; nvalid = 128; ct = g >> 2; kt = g & 3; }
  else                    { int g = f - FRAG_OUT; W = ow;  ldN = 32;  nvalid = 32;  ct = g >> 2; kt = g & 3; }
  int col = ct * 16 + (lane & 15);
  int hh  = lane >> 4;
  h16* dst = fb + (size_t)f * 512 + (size_t)lane * 16;
  for (int e = 0; e < 16; e++) {
    int k = kt * 32 + ((e < 8) ? (hh * 8 + e) : (16 + hh * 8 + (e - 8)));   // dense 16-bit B: table order, K 8h..8h+7 then 16+8h..16+8h+7
    float v = (col < nvalid) ? W[(size_t)k * ldN + col] : 0.0f;
    dst[e] = (h16)v;
  }
}

// ---------------- edge embed: RBF -> 64 -> 64 -> 8 (one wave per node) -------
__global__ __launch_bounds__(256) void edge_kernel(const float* edge_in,
    const float* b0, const float* b1, const float* b2,
    const h16* fb, h16* ee) {
  __shared__ h16   sx[8][16 * 64];
  __shared__ float sd[8][16];
  int lane = threadIdx.x & 31, wv = threadIdx.x >> 5;
  int node = blockIdx.x * 8 + wv;
  if (node >= Nn) return;
  int l16 = lane & 15, hh = lane >> 4;

  float d = edge_in[node * 16 + l16];
  if (hh == 0) sd[wv][l16] = d;
  float msk = d > 0.f ? 1.f : 0.f;
  const float gapc = (RBF_HIGH_F - RBF_LOW_F) / (float)(EH - 1);
  const float rgap = 1.f / gapc;

  // RBF expansion straight into A-fragment layout (K = 64 -> two ktiles)
  UA A0, A1;
#pragma unroll
  for (int e = 0; e < 16; e++) {
    int kl = (e < 8) ? (hh * 8 + e) : (16 + hh * 8 + (e - 8));
    float c0 = RBF_LOW_F + (float)kl * gapc;
    float df0 = d - c0;
    A0.h[e] = (h16)(__expf(-(df0 * df0) * rgap) * msk);
    float c1 = RBF_LOW_F + (float)(kl + 32) * gapc;
    float df1 = d - c1;
    A1.h[e] = (h16)(__expf(-(df1 * df1) * rgap) * msk);
  }

  // layer 1: [16x64] @ [64x64], relu
#pragma unroll
  for (int ct = 0; ct < 4; ct++) {
    UC acc{};
    acc.v = wmma16(A0.v, load_bfrag(fb, FRAG_W0 + ct * 2 + 0, lane), acc.v);
    acc.v = wmma16(A1.v, load_bfrag(fb, FRAG_W0 + ct * 2 + 1, lane), acc.v);
    float bb = b0[ct * 16 + l16];
#pragma unroll
    for (int v = 0; v < 8; v++) {
      float x = acc.f[v] + bb;
      x = x > 0.f ? x : 0.f;
      sx[wv][(v + 8 * hh) * 64 + ct * 16 + l16] = (h16)x;
    }
  }
  wbar();
  // layer 2
  v16h a0 = load_afrag(&sx[wv][0], 64, lane, 0);
  v16h a1 = load_afrag(&sx[wv][0], 64, lane, 1);
  wbar();
#pragma unroll
  for (int ct = 0; ct < 4; ct++) {
    UC acc{};
    acc.v = wmma16(a0, load_bfrag(fb, FRAG_W1 + ct * 2 + 0, lane), acc.v);
    acc.v = wmma16(a1, load_bfrag(fb, FRAG_W1 + ct * 2 + 1, lane), acc.v);
    float bb = b1[ct * 16 + l16];
#pragma unroll
    for (int v = 0; v < 8; v++) {
      float x = acc.f[v] + bb;
      x = x > 0.f ? x : 0.f;
      sx[wv][(v + 8 * hh) * 64 + ct * 16 + l16] = (h16)x;
    }
  }
  wbar();
  // layer 3: [16x64] @ [64x8] (zero-padded to 16 cols), tanh, mask
  a0 = load_afrag(&sx[wv][0], 64, lane, 0);
  a1 = load_afrag(&sx[wv][0], 64, lane, 1);
  UC acc{};
  acc.v = wmma16(a0, load_bfrag(fb, FRAG_W2 + 0, lane), acc.v);
  acc.v = wmma16(a1, load_bfrag(fb, FRAG_W2 + 1, lane), acc.v);
  if (l16 < 8) {
    float bb = b2[l16];
#pragma unroll
    for (int v = 0; v < 8; v++) {
      int r = v + 8 * hh;
      float m2 = sd[wv][r] > 0.f ? 1.f : 0.f;
      float x = fast_tanh(acc.f[v] + bb) * m2;
      ee[((size_t)node * 16 + r) * 8 + l16] = (h16)x;
    }
  }
}

// ---------------- message passing layer (one wave per 16-node tile) ----------
__global__ __launch_bounds__(128) void mp_kernel(const float* prev, float* next,
    const int* nlist, const h16* ee, const float* invd, const h16* fb, int layer) {
  __shared__ int snb[4][256];
  __shared__ U8  see[4][256];
  __shared__ h16 sT[4][16 * 256];
  int lane = threadIdx.x & 31, wv = threadIdx.x >> 5;
  int base = (blockIdx.x * 4 + wv) * 16;
  if (base >= Nn) return;

  // stage nlist + edge_embed into LDS via gfx1250 async global->LDS copies
  {
    unsigned long long gnb = (unsigned long long)(const void*)(nlist + (size_t)base * 16);
    unsigned long long gee = (unsigned long long)(const void*)(ee + (size_t)base * 16 * 8);
#pragma unroll
    for (int u = 0; u < 8; u++) {
      int idx = u * 32 + lane;
      unsigned int lnb = (unsigned int)(size_t)&snb[wv][idx];
      unsigned int lee = (unsigned int)(size_t)&see[wv][idx];
      unsigned long long anb = gnb + (size_t)idx * 4;
      unsigned long long aee = gee + (size_t)idx * 16;
      asm volatile("global_load_async_to_lds_b32 %0, %1, off"
                   :: "v"(lnb), "v"(anb) : "memory");
      asm volatile("global_load_async_to_lds_b128 %0, %1, off"
                   :: "v"(lee), "v"(aee) : "memory");
    }
    asm volatile("s_wait_asynccnt 0" ::: "memory");
  }
  wbar();

  // T[i, n, l] = sum_j g[i,j,n] * e[i,j,l]   (lane = feature n)
  for (int i = 0; i < 16; i++) {
    float acc[8] = {0, 0, 0, 0, 0, 0, 0, 0};
    for (int j = 0; j < 16; j++) {
      int nb = snb[wv][i * 16 + j];
      float g = prev[(size_t)nb * 32 + lane];
      U8 e = see[wv][i * 16 + j];
#pragma unroll
      for (int l = 0; l < 8; l++) acc[l] += g * (float)e.h[l];
    }
    U8 t;
#pragma unroll
    for (int l = 0; l < 8; l++) t.h[l] = (h16)acc[l];
    *(v4f*)(&sT[wv][i * 256 + lane * 8]) = t.f4;
  }
  wbar();

  // msg = T[16x256] @ W[256x32]
  UC c0{}, c1{};
#pragma unroll
  for (int kt = 0; kt < 8; kt++) {
    v16h a = load_afrag(&sT[wv][0], 256, lane, kt);
    c0.v = wmma16(a, load_bfrag(fb, FRAG_MP + layer * 16 + kt,     lane), c0.v);
    c1.v = wmma16(a, load_bfrag(fb, FRAG_MP + layer * 16 + 8 + kt, lane), c1.v);
  }

  int l16 = lane & 15, hh = lane >> 4;
#pragma unroll
  for (int ct = 0; ct < 2; ct++) {
    UC* c = ct ? &c1 : &c0;
    int m = ct * 16 + l16;
#pragma unroll
    for (int v = 0; v < 8; v++) {
      int r = base + v + 8 * hh;
      float x  = c->f[v] * invd[r];
      next[(size_t)r * 32 + m] = fast_softplus(x) + prev[(size_t)r * 32 + m];
    }
  }
}

// ---------------- FC block + peak reduction (one wave per 16-node tile) ------
__global__ __launch_bounds__(256) void fc_kernel(const float* nodes,
    const float* node_in, const float* fb0, const float* fb1, const float* fb2,
    const float* ob, const float* pstd, const float* pavg,
    const h16* fb, float* out) {
  __shared__ h16 sh[8][16 * 128];
  int lane = threadIdx.x & 31, wv = threadIdx.x >> 5;
  int base = (blockIdx.x * 8 + wv) * 16;
  if (base >= Nn) return;
  int l16 = lane & 15, hh = lane >> 4;

  for (int r = 0; r < 16; r++)
    sh[wv][r * 128 + lane] = (h16)nodes[(size_t)(base + r) * 32 + lane];
  wbar();

  // fc0: 32 -> 128, relu
  {
    v16h a = load_afrag(&sh[wv][0], 128, lane, 0);
    wbar();
#pragma unroll
    for (int ct = 0; ct < 8; ct++) {
      UC acc{};
      acc.v = wmma16(a, load_bfrag(fb, FRAG_FC0 + ct, lane), acc.v);
      float bb = fb0[ct * 16 + l16];
#pragma unroll
      for (int v = 0; v < 8; v++) {
        float x = acc.f[v] + bb;
        x = x > 0.f ? x : 0.f;
        sh[wv][(v + 8 * hh) * 128 + ct * 16 + l16] = (h16)x;
      }
    }
  }
  wbar();
  // fc1: 128 -> 128, relu
  {
    v16h a[4];
#pragma unroll
    for (int kt = 0; kt < 4; kt++) a[kt] = load_afrag(&sh[wv][0], 128, lane, kt);
    wbar();
#pragma unroll
    for (int ct = 0; ct < 8; ct++) {
      UC acc{};
#pragma unroll
      for (int kt = 0; kt < 4; kt++)
        acc.v = wmma16(a[kt], load_bfrag(fb, FRAG_FC1 + ct * 4 + kt, lane), acc.v);
      float bb = fb1[ct * 16 + l16];
#pragma unroll
      for (int v = 0; v < 8; v++) {
        float x = acc.f[v] + bb;
        x = x > 0.f ? x : 0.f;
        sh[wv][(v + 8 * hh) * 128 + ct * 16 + l16] = (h16)x;
      }
    }
  }
  wbar();
  // fc2: 128 -> 128, tanh
  {
    v16h a[4];
#pragma unroll
    for (int kt = 0; kt < 4; kt++) a[kt] = load_afrag(&sh[wv][0], 128, lane, kt);
    wbar();
#pragma unroll
    for (int ct = 0; ct < 8; ct++) {
      UC acc{};
#pragma unroll
      for (int kt = 0; kt < 4; kt++)
        acc.v = wmma16(a[kt], load_bfrag(fb, FRAG_FC2 + ct * 4 + kt, lane), acc.v);
      float bb = fb2[ct * 16 + l16];
#pragma unroll
      for (int v = 0; v < 8; v++) {
        float x = fast_tanh(acc.f[v] + bb);
        sh[wv][(v + 8 * hh) * 128 + ct * 16 + l16] = (h16)x;
      }
    }
  }
  wbar();
  // out: 128 -> 32, then masked sum over 32 cols -> peaks
  {
    v16h a[4];
#pragma unroll
    for (int kt = 0; kt < 4; kt++) a[kt] = load_afrag(&sh[wv][0], 128, lane, kt);
    float pe[8] = {0, 0, 0, 0, 0, 0, 0, 0};
#pragma unroll
    for (int ct = 0; ct < 2; ct++) {
      UC acc{};
#pragma unroll
      for (int kt = 0; kt < 4; kt++)
        acc.v = wmma16(a[kt], load_bfrag(fb, FRAG_OUT + ct * 4 + kt, lane), acc.v);
      int m = ct * 16 + l16;
      float sm = pstd[m], am = pavg[m], bm = ob[m];
#pragma unroll
      for (int v = 0; v < 8; v++) {
        int r = base + v + 8 * hh;
        float ni = node_in[(size_t)r * 32 + m];
        pe[v] += (acc.f[v] + bm) * ni * sm + ni * am;
      }
    }
#pragma unroll
    for (int off = 1; off < 16; off <<= 1) {
#pragma unroll
      for (int v = 0; v < 8; v++) pe[v] += __shfl_xor(pe[v], off, 32);
    }
    if (l16 == 0) {
#pragma unroll
      for (int v = 0; v < 8; v++) out[base + v + 8 * hh] = pe[v];
    }
  }
}

// ---------------- host launcher ----------------
extern "C" void kernel_launch(void* const* d_in, const int* in_sizes, int n_in,
                              void* d_out, int out_size, void* d_ws, size_t ws_size,
                              hipStream_t stream) {
  (void)in_sizes; (void)n_in; (void)out_size; (void)ws_size;
  const float* node_in = (const float*)d_in[0];
  const int*   nlist   = (const int*)d_in[1];
  const float* edge_in = (const float*)d_in[2];
  const float* invd    = (const float*)d_in[3];
  const float* ew0 = (const float*)d_in[4];
  const float* ew1 = (const float*)d_in[5];
  const float* ew2 = (const float*)d_in[6];
  const float* eb0 = (const float*)d_in[7];
  const float* eb1 = (const float*)d_in[8];
  const float* eb2 = (const float*)d_in[9];
  const float* mp0 = (const float*)d_in[10];
  const float* mp1 = (const float*)d_in[11];
  const float* mp2 = (const float*)d_in[12];
  const float* mp3 = (const float*)d_in[13];
  const float* fw0 = (const float*)d_in[14];
  const float* fw1 = (const float*)d_in[15];
  const float* fw2 = (const float*)d_in[16];
  const float* fcb0 = (const float*)d_in[17];
  const float* fcb1 = (const float*)d_in[18];
  const float* fcb2 = (const float*)d_in[19];
  const float* outw = (const float*)d_in[20];
  const float* outb = (const float*)d_in[21];
  const float* pstd = (const float*)d_in[22];
  const float* pavg = (const float*)d_in[23];

  char* ws = (char*)d_ws;
  h16* fragbuf = (h16*)ws;
  size_t off = (size_t)NFRAGS * 1024;                 // 162 KB, 256B-aligned
  h16* ee = (h16*)(ws + off);                         // [N,16,8] f16
  off += (size_t)Nn * 16 * 8 * sizeof(h16);
  float* nodes0 = (float*)(ws + off);                 // [N,32] f32 ping
  off += (size_t)Nn * 32 * sizeof(float);
  float* nodes1 = (float*)(ws + off);                 // [N,32] f32 pong

  prep_frags<<<(NFRAGS * 32 + 255) / 256, 256, 0, stream>>>(
      ew0, ew1, ew2, mp0, mp1, mp2, mp3, fw0, fw1, fw2, outw, fragbuf);

  edge_kernel<<<(Nn + 7) / 8, 256, 0, stream>>>(edge_in, eb0, eb1, eb2, fragbuf, ee);

  int tiles = (Nn + 15) / 16;          // 6250 sixteen-node tiles
  int mpb   = (tiles + 3) / 4;         // 4 waves / 128 threads per block
  mp_kernel<<<mpb, 128, 0, stream>>>(node_in, nodes0, nlist, ee, invd, fragbuf, 0);
  mp_kernel<<<mpb, 128, 0, stream>>>(nodes0, nodes1, nlist, ee, invd, fragbuf, 1);
  mp_kernel<<<mpb, 128, 0, stream>>>(nodes1, nodes0, nlist, ee, invd, fragbuf, 2);
  mp_kernel<<<mpb, 128, 0, stream>>>(nodes0, nodes1, nlist, ee, invd, fragbuf, 3);

  fc_kernel<<<(tiles + 7) / 8, 256, 0, stream>>>(
      nodes1, node_in, fcb0, fcb1, fcb2, outb, pstd, pavg, fragbuf, (float*)d_out);
}
